// LGCN_16518444220730
// MI455X (gfx1250) — matrix operand order repaired and checked by reading, not compiled, not run on hardware
//
#include <hip/hip_runtime.h>
#include <hip/hip_bf16.h>

typedef __attribute__((ext_vector_type(16))) _Float16 v16h;
typedef __attribute__((ext_vector_type(8)))  float    v8f;

#define DIM 64

// ---------------- utility kernels ----------------

__global__ void k_zero(float* __restrict__ p, size_t n) {
    size_t i = (size_t)blockIdx.x * blockDim.x + threadIdx.x;
    if (i < n) p[i] = 0.0f;
}

__global__ void k_degree(const int* __restrict__ col, float* __restrict__ deg, size_t E) {
    size_t i = (size_t)blockIdx.x * blockDim.x + threadIdx.x;
    if (i < E) atomicAdd(&deg[col[i]], 1.0f);
}

__global__ void k_rsqrt(float* __restrict__ d, int n) {
    int i = blockIdx.x * blockDim.x + threadIdx.x;
    if (i < n) {
        float v = d[i];
        d[i] = (v > 0.0f) ? rsqrtf(v) : 0.0f;
    }
}

// x = acc = concat(user_emb, item_emb)
__global__ void k_init(const float* __restrict__ ue, const float* __restrict__ ie,
                       float* __restrict__ x, float* __restrict__ acc,
                       size_t userElems, size_t total) {
    size_t i = (size_t)blockIdx.x * blockDim.x + threadIdx.x;
    if (i < total) {
        float v = (i < userElems) ? ue[i] : ie[i - userElems];
        x[i]   = v;
        acc[i] = v;
    }
}

// One wave per edge; each lane handles 2 of the 64 features.
// xn[row] += edge_val * dinv[row]*dinv[col] * x[col]
__global__ void k_spmm(const float* __restrict__ x, float* __restrict__ xn,
                       const int* __restrict__ row, const int* __restrict__ col,
                       const float* __restrict__ val, const float* __restrict__ dinv,
                       size_t E) {
    size_t gtid = (size_t)blockIdx.x * blockDim.x + threadIdx.x;
    size_t e = gtid >> 5;
    int lane = threadIdx.x & 31;
    if (e >= E) return;
    int r = row[e];
    int c = col[e];
    float w = val[e] * dinv[r] * dinv[c];
    const float2 xv = *(const float2*)(x + (size_t)c * DIM + lane * 2);
    float* dst = xn + (size_t)r * DIM + lane * 2;
    atomicAdd(dst,     w * xv.x);
    atomicAdd(dst + 1, w * xv.y);
}

__global__ void k_add(float* __restrict__ acc, const float* __restrict__ xn, size_t n) {
    size_t i = (size_t)blockIdx.x * blockDim.x + threadIdx.x;
    if (i < n) acc[i] += xn[i];
}

// o[r][d] = f16( g[idx[r]][d] * 0.25 )
__global__ void k_gather_f16(const float* __restrict__ g, const int* __restrict__ idx,
                             _Float16* __restrict__ o, size_t total) {
    size_t i = (size_t)blockIdx.x * blockDim.x + threadIdx.x;
    if (i < total) {
        size_t r = i >> 6;
        int d = (int)(i & 63);
        o[i] = (_Float16)(g[(size_t)idx[r] * DIM + d] * 0.25f);
    }
}

// ---------------- WMMA GEMM: out(M x Ncols) = A(M x 64) * B(Ncols x 64)^T ----------------
// 8 waves / block. Each wave computes one 16x16 output tile via two
// v_wmma_f32_16x16x32_f16 (K = 64). All 8 waves in a block share one N-tile
// of B (L2/L0 reuse); A (512x64) is trivially cache resident.

__global__ __launch_bounds__(256)
void k_gemm(const _Float16* __restrict__ A, const _Float16* __restrict__ Bm,
            float* __restrict__ out, int M, int Ncols, int ntiles) {
    const int lane = threadIdx.x & 31;
    const int wave = threadIdx.x >> 5;
    const int ntile  = blockIdx.x % ntiles;
    const int mchunk = blockIdx.x / ntiles;
    const int m0 = (mchunk * 8 + wave) * 16;
    const int n0 = ntile * 16;
    if (m0 >= M) return;                 // wave-uniform: EXEC stays all-1s

    const int lr = lane & 15;
    const int hi = lane >> 4;

    int arow = m0 + lr; if (arow >= M) arow = M - 1;
    int bcol = n0 + lr; if (bcol >= Ncols) bcol = Ncols - 1;
    const unsigned int* Ar = (const unsigned int*)(A  + (size_t)arow * DIM); // 32 dwords
    const unsigned int* Br = (const unsigned int*)(Bm + (size_t)bcol * DIM);

    union UF { v16h h; unsigned int u[8]; };
    union CF { v8f  v; float        f[8]; };

    CF c;
#pragma unroll
    for (int i = 0; i < 8; ++i) c.f[i] = 0.0f;

#pragma unroll
    for (int kk = 0; kk < 2; ++kk) {
        const int k0 = kk * 32;
        UF a, b;
#pragma unroll
        for (int j = 0; j < 8; ++j) {
            // A 16x32 f16 layout: lanes 0-15 -> K {2j,2j+1} (j<4), {16+2(j-4)} (j>=4)
            //                     lanes 16-31 -> +8 within each K-octet group
            const int grp = j >> 2, w = j & 3;
            const int ka = k0 + grp * 16 + hi * 8 + w * 2;
            a.u[j] = Ar[ka >> 1];
            // B 32x16 f16 layout: same K mapping as A
            const int kb = k0 + grp * 16 + hi * 8 + w * 2;
            b.u[j] = Br[kb >> 1];
        }
        c.v = __builtin_amdgcn_wmma_f32_16x16x32_f16(
                  false, a.h, false, b.h, (short)0, c.v, false, false);
    }

    // C/D f32 16x16: VGPR v -> M = v + 8*hi, N = lane&15
    const int nc = n0 + lr;
    if (m0 + 16 <= M && n0 + 16 <= Ncols) {
        // Full tile (always taken for 512 x 50000): unpredicated strided stores.
        float* p = out + (size_t)(m0 + hi * 8) * (size_t)Ncols + nc;
#pragma unroll
        for (int v = 0; v < 8; ++v) {
            *p = c.f[v];
            p += Ncols;
        }
    } else if (nc < Ncols) {
        // Ragged tail path (generic safety; never taken for these shapes).
        const int mbase = m0 + hi * 8;
#pragma unroll
        for (int v = 0; v < 8; ++v) {
            const int m = mbase + v;
            if (m < M) out[(size_t)m * Ncols + nc] = c.f[v];
        }
    }
}

// ---------------- host orchestration ----------------

extern "C" void kernel_launch(void* const* d_in, const int* in_sizes, int n_in,
                              void* d_out, int out_size, void* d_ws, size_t ws_size,
                              hipStream_t stream) {
    const float* user_emb = (const float*)d_in[0];
    const float* item_emb = (const float*)d_in[1];
    const float* edge_val = (const float*)d_in[2];
    const int*   edge_row = (const int*)d_in[3];
    const int*   edge_col = (const int*)d_in[4];
    const int*   user_idx = (const int*)d_in[5];
    // d_in[6] item_seq_indices: unused by the reference output
    const int*   tgt_idx  = (const int*)d_in[7];

    const int    NU = in_sizes[0] / DIM;
    const int    NI = in_sizes[1] / DIM;
    const int    Nn = NU + NI;
    const size_t E  = (size_t)in_sizes[2];
    const int    Mb = in_sizes[5];   // 512 users scored
    const int    Cc = in_sizes[7];   // 50000 targets

    // workspace carve-out (256B aligned): dinv, x ping/pong, acc, f16 A/B
    char* ws = (char*)d_ws;
    size_t off = 0;
    auto take = [&](size_t bytes) -> char* {
        char* p = ws + off;
        off += (bytes + 255) & ~((size_t)255);
        return p;
    };
    float*    dinv = (float*)take((size_t)Nn * 4);
    float*    xa   = (float*)take((size_t)Nn * DIM * 4);
    float*    xb   = (float*)take((size_t)Nn * DIM * 4);
    float*    acc  = (float*)take((size_t)Nn * DIM * 4);
    _Float16* Af   = (_Float16*)take((size_t)Mb * DIM * 2);
    _Float16* Bf   = (_Float16*)take((size_t)Cc * DIM * 2);
    (void)ws_size; (void)n_in; (void)out_size;

    const int T = 256;
    auto nb = [](size_t n, int t) { return (unsigned)((n + (size_t)t - 1) / (size_t)t); };
    const size_t nodeElems = (size_t)Nn * DIM;

    // normalization: deg over edge_col, then rsqrt
    k_zero  <<<nb(Nn, T), T, 0, stream>>>(dinv, (size_t)Nn);
    k_degree<<<nb(E, T),  T, 0, stream>>>(edge_col, dinv, E);
    k_rsqrt <<<nb(Nn, T), T, 0, stream>>>(dinv, Nn);

    // x = acc = concat(user_emb, item_emb)
    k_init<<<nb(nodeElems, T), T, 0, stream>>>(user_emb, item_emb, xa, acc,
                                               (size_t)NU * DIM, nodeElems);

    float* x  = xa;
    float* xn = xb;
    for (int l = 0; l < 3; ++l) {
        k_zero<<<nb(nodeElems, T), T, 0, stream>>>(xn, nodeElems);
        k_spmm<<<nb(E * 32, T), T, 0, stream>>>(x, xn, edge_row, edge_col,
                                                edge_val, dinv, E);
        k_add <<<nb(nodeElems, T), T, 0, stream>>>(acc, xn, nodeElems);
        float* t = x; x = xn; xn = t;
    }

    // gather + /4 + f16 convert
    k_gather_f16<<<nb((size_t)Mb * DIM, T), T, 0, stream>>>(acc, user_idx, Af,
                                                            (size_t)Mb * DIM);
    k_gather_f16<<<nb((size_t)Cc * DIM, T), T, 0, stream>>>(acc, tgt_idx, Bf,
                                                            (size_t)Cc * DIM);

    // WMMA GEMM: 512 x 50000, K=64
    const int ntiles  = (Cc + 15) / 16;
    const int mtiles  = (Mb + 15) / 16;
    const int mchunks = (mtiles + 7) / 8;
    k_gemm<<<(unsigned)(ntiles * mchunks), 256, 0, stream>>>(Af, Bf, (float*)d_out,
                                                             Mb, Cc, ntiles);
}
